// Temporal_Attn_Head_90847148245610
// MI455X (gfx1250) — matrix operand order repaired and checked by reading, not compiled, not run on hardware
//
#include <hip/hip_runtime.h>
#include <hip/hip_bf16.h>

typedef __attribute__((ext_vector_type(2)))  float  v2f;
typedef __attribute__((ext_vector_type(8)))  float  v8f;
typedef __attribute__((ext_vector_type(16))) __bf16 v16bf;

#define NN 8192
#define CC 256
#define DD 64
#define JSPLIT 8
#define ISPLIT 8

__device__ __forceinline__ __bf16 f2bf(float f) {
  unsigned u = __float_as_uint(f);
  unsigned r = u + 0x7FFFu + ((u >> 16) & 1u);  // round-to-nearest-even
  unsigned short hb = (unsigned short)(r >> 16);
  __bf16 o;
  __builtin_memcpy(&o, &hb, 2);
  return o;
}

// ---------------------------------------------------------------------------
// k0: h = x @ W1^T via V_WMMA_F32_16X16X4_F32 (full f32 precision).
// Block = 128 threads = 4 waves; wave = d-tile, block = 16 rows of x.
// Writes h (f32, N x D) and hT (bf16, D x N) for k3's contiguous B fragments.
// ---------------------------------------------------------------------------
__global__ __launch_bounds__(128)
void k0_h(const float* __restrict__ x, const float* __restrict__ W1,
          float* __restrict__ h, __bf16* __restrict__ hT) {
  const int wave  = threadIdx.x >> 5;
  const int lane  = threadIdx.x & 31;
  const int m16   = lane & 15;
  const int khalf = (lane >> 4) * 2;          // 32-bit A/B frag: lanes 16-31 hold K=2,3
  const int n0    = blockIdx.x * 16;
  const int dcol  = wave * 16 + m16;

  const float* xrow = x  + (size_t)(n0 + m16) * CC;
  const float* wrow = W1 + (size_t)dcol * CC; // B[k][n] = W1[n][k]

  v8f acc = {};
  #pragma unroll 4
  for (int k0 = 0; k0 < CC; k0 += 4) {
    v2f a = *(const v2f*)(xrow + k0 + khalf);
    v2f b = *(const v2f*)(wrow + k0 + khalf);
    acc = __builtin_amdgcn_wmma_f32_16x16x4_f32(false, a, false, b,
                                                (short)0, acc, false, false);
  }

  const int rbase = n0 + ((lane >> 4) << 3);
  #pragma unroll
  for (int r = 0; r < 8; ++r) {
    float v = acc[r];
    h [(size_t)(rbase + r) * DD + dcol] = v;
    hT[(size_t)dcol * NN + (rbase + r)] = f2bf(v);
  }
}

// ---------------------------------------------------------------------------
// k1: f1[n] = h[n,:]·a1 ; f2[n] = h[n,:]·a2
// ---------------------------------------------------------------------------
__global__ __launch_bounds__(256)
void k1_f(const float* __restrict__ h, const float* __restrict__ a1,
          const float* __restrict__ a2, float* __restrict__ f1,
          float* __restrict__ f2) {
  int n = blockIdx.x * 256 + threadIdx.x;
  const float* hr = h + (size_t)n * DD;
  float s1 = 0.f, s2 = 0.f;
  #pragma unroll 8
  for (int d = 0; d < DD; ++d) {
    float hv = hr[d];
    s1 += hv * a1[d];
    s2 += hv * a2[d];
  }
  f1[n] = s1;
  f2[n] = s2;
}

// ---------------------------------------------------------------------------
// k2: per-column online-softmax partial stats over an i-slice.
// Thread = column j; blockIdx.y selects i-range slice. Coalesced bias reads,
// f2 staged via LDS, branchy online update (1 exp in the common path).
// ---------------------------------------------------------------------------
__global__ __launch_bounds__(256)
void k2_stats(const float* __restrict__ bias, const float* __restrict__ f1,
              const float* __restrict__ f2, float2* __restrict__ pstat) {
  __shared__ float sf2[256];
  const int j    = blockIdx.x * 256 + threadIdx.x;
  const int ibeg = blockIdx.y * (NN / ISPLIT);
  const int iend = ibeg + (NN / ISPLIT);
  const float f1j = f1[j];
  float m = -3.0e38f, s = 0.f;
  for (int i0 = ibeg; i0 < iend; i0 += 256) {
    sf2[threadIdx.x] = f2[i0 + threadIdx.x];
    __syncthreads();
    if (i0 + 256 < iend)
      __builtin_prefetch(bias + (size_t)(i0 + 256) * NN + j, 0, 1);
    #pragma unroll 4
    for (int ii = 0; ii < 256; ++ii) {
      float z = sf2[ii] + f1j;
      float L = fmaxf(z, 0.01f * z) + bias[(size_t)(i0 + ii) * NN + j];
      if (L > m) { s = s * __expf(m - L) + 1.0f; m = L; }
      else       { s += __expf(L - m); }
    }
    __syncthreads();
  }
  pstat[(size_t)blockIdx.y * NN + j] = make_float2(m, s);
}

// ---------------------------------------------------------------------------
// k2b: merge ISPLIT online-softmax partials; emit jp[j] = (f1, M, 1/S, t)
// ---------------------------------------------------------------------------
__global__ __launch_bounds__(256)
void k2_merge(const float2* __restrict__ pstat, const float* __restrict__ f1,
              const float* __restrict__ bt, float4* __restrict__ jp) {
  int j = blockIdx.x * 256 + threadIdx.x;
  float2 v[ISPLIT];
  float m = -3.0e38f;
  #pragma unroll
  for (int p = 0; p < ISPLIT; ++p) {
    v[p] = pstat[(size_t)p * NN + j];
    m = fmaxf(m, v[p].x);
  }
  float s = 0.f;
  #pragma unroll
  for (int p = 0; p < ISPLIT; ++p) s += v[p].y * __expf(v[p].x - m);
  jp[j] = make_float4(f1[j], m, 1.0f / s, bt[j]);
}

// ---------------------------------------------------------------------------
// k3: partial[i,:] += sum_j exp(leaky(f2_i+f1_j)+bias_ij - M_j - lam|t_i-t_j|)
//                     * invS_j * h[j,:]      via V_WMMA_F32_16X16X32_BF16.
// Block = 128 thr = 4 waves; wave owns 16 rows x 64 cols (4 accumulators).
// j-range split JSPLIT ways over blockIdx.y; jp slice staged in LDS (16 KB,
// broadcast ds_load_b128 reads). A fragment built in contiguous K order (k = 16*half + e), matching B.
// ---------------------------------------------------------------------------
__global__ __launch_bounds__(128)
void k3_attn(const float* __restrict__ bias, const float4* __restrict__ jp,
             const float* __restrict__ f2, const float* __restrict__ bt,
             const __bf16* __restrict__ hT, const float* __restrict__ lam_p,
             float* __restrict__ partial) {
  __shared__ float4 sjp[NN / JSPLIT];  // 1024 * 16B = 16 KB
  const int wave = threadIdx.x >> 5;
  const int lane = threadIdx.x & 31;
  const int m16  = lane & 15;
  const int half = lane >> 4;
  const int kb   = half * 8;
  const int i    = blockIdx.x * 64 + wave * 16 + m16;

  const int jbeg = blockIdx.y * (NN / JSPLIT);

  for (int idx = threadIdx.x; idx < NN / JSPLIT; idx += 128)
    sjp[idx] = jp[jbeg + idx];
  __syncthreads();

  const float lam = *lam_p;
  const float fi2 = f2[i];
  const float ti  = bt[i];

  v8f acc0 = {}, acc1 = {}, acc2 = {}, acc3 = {};
  const float* brow = bias + (size_t)i * NN + jbeg;

  for (int jo = 0; jo < NN / JSPLIT; jo += 32) {
    if (jo + 64 < NN / JSPLIT)
      __builtin_prefetch(brow + jo + 64 + kb, 0, 1);

    // bias for this lane's 16 (i,j) elements: one run of 16 consecutive j
    const float4* bias4 = (const float4*)(brow + jo);
    float4 b0 = bias4[(kb >> 1) + 0];
    float4 b1 = bias4[(kb >> 1) + 1];
    float4 b2 = bias4[(kb >> 1) + 2];
    float4 b3 = bias4[(kb >> 1) + 3];
    float bv[16] = {b0.x, b0.y, b0.z, b0.w, b1.x, b1.y, b1.z, b1.w,
                    b2.x, b2.y, b2.z, b2.w, b3.x, b3.y, b3.z, b3.w};

    v16bf a;
    #pragma unroll
    for (int e = 0; e < 16; ++e) {
      int k = half * 16 + e;                       // contiguous K mapping (matches B)
      float4 p = sjp[jo + k];                      // (f1, M, invS, t)
      float z  = fi2 + p.x;
      float L  = fmaxf(z, 0.01f * z) + bv[e];      // leaky_relu + bias
      float ex = __expf(L - p.y - lam * fabsf(ti - p.w));
      a[e] = f2bf(ex * p.z);
    }

    // B frags: lane holds col n=m16, K = half*16 + 0..15 (contiguous bf16)
    const __bf16* hb = hT + jbeg + jo + half * 16;
    v16bf B0 = *(const v16bf*)(hb + (size_t)(0 * 16 + m16) * NN);
    v16bf B1 = *(const v16bf*)(hb + (size_t)(1 * 16 + m16) * NN);
    v16bf B2 = *(const v16bf*)(hb + (size_t)(2 * 16 + m16) * NN);
    v16bf B3 = *(const v16bf*)(hb + (size_t)(3 * 16 + m16) * NN);

    acc0 = __builtin_amdgcn_wmma_f32_16x16x32_bf16(false, a, false, B0, (short)0, acc0, false, false);
    acc1 = __builtin_amdgcn_wmma_f32_16x16x32_bf16(false, a, false, B1, (short)0, acc1, false, false);
    acc2 = __builtin_amdgcn_wmma_f32_16x16x32_bf16(false, a, false, B2, (short)0, acc2, false, false);
    acc3 = __builtin_amdgcn_wmma_f32_16x16x32_bf16(false, a, false, B3, (short)0, acc3, false, false);
  }

  float* pp = partial + (size_t)blockIdx.y * (NN * DD);
  const int rbase = blockIdx.x * 64 + wave * 16 + half * 8;
  #pragma unroll
  for (int r = 0; r < 8; ++r) {
    float* row = pp + (size_t)(rbase + r) * DD + m16;
    row[0]  = acc0[r];
    row[16] = acc1[r];
    row[32] = acc2[r];
    row[48] = acc3[r];
  }
}

// ---------------------------------------------------------------------------
// k4: reduce JSPLIT partials + ELU
// ---------------------------------------------------------------------------
__global__ __launch_bounds__(256)
void k4_reduce(const float* __restrict__ partial, float* __restrict__ out) {
  int idx = blockIdx.x * 256 + threadIdx.x;
  float s = 0.f;
  #pragma unroll
  for (int p = 0; p < JSPLIT; ++p) s += partial[(size_t)p * (NN * DD) + idx];
  out[idx] = (s > 0.f) ? s : (__expf(s) - 1.0f);
}

extern "C" void kernel_launch(void* const* d_in, const int* in_sizes, int n_in,
                              void* d_out, int out_size, void* d_ws, size_t ws_size,
                              hipStream_t stream) {
  (void)in_sizes; (void)n_in; (void)out_size; (void)ws_size;
  const float* x    = (const float*)d_in[0];
  const float* bias = (const float*)d_in[1];
  const float* bt   = (const float*)d_in[2];
  const float* lam  = (const float*)d_in[3];
  const float* W1   = (const float*)d_in[4];
  const float* a1   = (const float*)d_in[5];
  const float* a2   = (const float*)d_in[6];
  float* out = (float*)d_out;

  char* ws = (char*)d_ws;
  float*  h       = (float*)ws;   ws += (size_t)NN * DD * 4;       // 2 MB
  __bf16* hT      = (__bf16*)ws;  ws += (size_t)DD * NN * 2;       // 1 MB
  float*  f1      = (float*)ws;   ws += (size_t)NN * 4;
  float*  f2v     = (float*)ws;   ws += (size_t)NN * 4;
  float4* jp      = (float4*)ws;  ws += (size_t)NN * 16;
  float2* pstat   = (float2*)ws;  ws += (size_t)ISPLIT * NN * 8;   // 512 KB
  float*  partial = (float*)ws;                                    // JSPLIT * 2 MB

  k0_h    <<<NN / 16,  128, 0, stream>>>(x, W1, h, hT);
  k1_f    <<<NN / 256, 256, 0, stream>>>(h, a1, a2, f1, f2v);
  dim3 g2(NN / 256, ISPLIT);
  k2_stats<<<g2, 256, 0, stream>>>(bias, f1, f2v, pstat);
  k2_merge<<<NN / 256, 256, 0, stream>>>(pstat, f1, bt, jp);
  dim3 g3(NN / 64, JSPLIT);
  k3_attn <<<g3, 128, 0, stream>>>(bias, jp, f2v, bt, hT, lam, partial);
  k4_reduce<<<(NN * DD) / 256, 256, 0, stream>>>(partial, out);
}
